// GNN_17076789969486
// MI455X (gfx1250) — compile-verified, matrix operand order checked by reading, not run on hardware
//
#include <hip/hip_runtime.h>

// ---------------- constants matching the reference ----------------
#define F_IN   64
#define HC     128      // H * C
#define NHEAD  4
#define CHAN   32
#define NCLS   16
#define SLOPE  0.2f

typedef __attribute__((ext_vector_type(2)))  float    v2f;
typedef __attribute__((ext_vector_type(8)))  float    v8f;
typedef __attribute__((ext_vector_type(16))) _Float16 v16h;

// ---------------- helpers ----------------
__device__ __forceinline__ unsigned fkey(float f) {
    unsigned u = __float_as_uint(f);
    return (u & 0x80000000u) ? ~u : (u | 0x80000000u);
}
__device__ __forceinline__ float funkey(unsigned k) {
    unsigned u = (k & 0x80000000u) ? (k & 0x7FFFFFFFu) : ~k;
    return __uint_as_float(u);
}

__global__ void k_zero(float* __restrict__ p, long long n) {
    long long i = (long long)blockIdx.x * blockDim.x + threadIdx.x;
    long long stride = (long long)gridDim.x * blockDim.x;
    for (; i < n; i += stride) p[i] = 0.0f;
}

// ---------------- WMMA GEMM:  OUT[N,128] = X[N,din] @ W[din,128] + b ----------------
// grid.x = ceil(N/16) row tiles, grid.y = 2 (lin_l / lin_r). block = 256 (8 waves);
// the 16 x din A tile is staged once into LDS (stride din+4 -> conflict-free),
// then all 8 waves (one 16x16 n-tile each) feed their A fragments from LDS.
#define LDS_STRIDE_PAD 4
__global__ void k_gemm(const float* __restrict__ X,
                       const float* __restrict__ Wl, const float* __restrict__ bl,
                       const float* __restrict__ Wr, const float* __restrict__ br,
                       float* __restrict__ XL, float* __restrict__ XR,
                       int Nn, int din) {
    __shared__ float Atile[16 * (HC + LDS_STRIDE_PAD)];

    const float* W    = blockIdx.y ? Wr : Wl;
    const float* bias = blockIdx.y ? br : bl;
    float*       OUT  = blockIdx.y ? XR : XL;

    const int wave = threadIdx.x >> 5;
    const int lane = threadIdx.x & 31;
    const int hf   = lane >> 4;      // half-wave: 0 or 1
    const int l    = lane & 15;
    const int m0   = blockIdx.x * 16;
    const int n0   = wave * 16;
    const int ldsStride = din + LDS_STRIDE_PAD;

    // cooperative A-tile stage: 16*din elements over 256 threads
    const int elems = 16 * din;
    for (int idx = threadIdx.x; idx < elems; idx += 256) {
        const int row = idx / din;
        const int col = idx - row * din;
        const int grow = m0 + row;
        Atile[row * ldsStride + col] =
            (grow < Nn) ? X[(size_t)grow * din + col] : 0.0f;
    }
    __syncthreads();

    v8f acc = {0.f, 0.f, 0.f, 0.f, 0.f, 0.f, 0.f, 0.f};
    const float* Arow = &Atile[l * ldsStride];

#if __has_builtin(__builtin_amdgcn_wmma_f32_16x16x4_f32)
    // fp32 path: V_WMMA_F32_16X16X4_F32.
    // A 16x4: lanes 0-15 rows, K={0,1}; lanes 16-31 rows, K={2,3}.
    // B 4x16: lanes hold column n=l; same K split across halves.
    for (int k = 0; k < din; k += 4) {
        const int ka = k + 2 * hf;
        v2f a;
        a.x = Arow[ka];
        a.y = Arow[ka + 1];
        v2f b;
        b.x = W[(size_t)ka       * HC + n0 + l];
        b.y = W[(size_t)(ka + 1) * HC + n0 + l];
        acc = __builtin_amdgcn_wmma_f32_16x16x4_f32(
                  false, a, false, b, (short)0, acc, false, false);
    }
#else
    // fallback: f16 inputs, fp32 accumulate (codegen-confirmed builtin)
    for (int k = 0; k < din; k += 32) {
        v16h a, b;
#pragma unroll
        for (int v = 0; v < 8; ++v) {
            const int kb = ((v < 4) ? 2 * v : 16 + 2 * (v - 4)) + (hf ? 8 : 0) + k;
            a[2 * v]     = (_Float16)Arow[kb];
            a[2 * v + 1] = (_Float16)Arow[kb + 1];
            b[2 * v]     = (_Float16)(W[(size_t)kb       * HC + n0 + l]);
            b[2 * v + 1] = (_Float16)(W[(size_t)(kb + 1) * HC + n0 + l]);
        }
        acc = __builtin_amdgcn_wmma_f32_16x16x32_f16(
                  false, a, false, b, (short)0, acc, false, false);
    }
#endif

    const int   col = n0 + l;
    const float bv  = bias[col];
#pragma unroll
    for (int j = 0; j < 8; ++j) {
        const int r = m0 + j + 8 * hf;   // C/D layout: VGPR j -> row j + 8*half
        if (r < Nn) OUT[(size_t)r * HC + col] = acc[j] + bv;
    }
}

// ---------------- edge attention score + segment max ----------------
// 8 lanes per (edge, head): lane q owns channels [4q, 4q+4) -> each 8-lane group
// issues one contiguous 128B gather; dot-product reduced with a shfl_xor butterfly.
__global__ void k_score(const int* __restrict__ ei, int E, int Nn,
                        const float* __restrict__ XL, const float* __restrict__ XR,
                        const float* __restrict__ att,
                        float* __restrict__ earr, unsigned* __restrict__ emaxk) {
    const long long tid = (long long)blockIdx.x * blockDim.x + threadIdx.x;
    const long long tot = (long long)(E + Nn) * NHEAD * 8;
    if (tid >= tot) return;
    const int gid = (int)(tid >> 3);   // (e, h) pair
    const int q   = (int)(tid & 7);    // channel quad
    const int e = gid >> 2, h = gid & 3;
    int src, dst;
    if (e < E) { src = ei[e]; dst = ei[E + e]; } else { src = dst = e - E; }

    const float4 a = *(const float4*)(XL + (size_t)src * HC + h * CHAN + q * 4);
    const float4 b = *(const float4*)(XR + (size_t)dst * HC + h * CHAN + q * 4);
    const float4 w = *(const float4*)(att + h * CHAN + q * 4);

    float mx = a.x + b.x, my = a.y + b.y, mz = a.z + b.z, mw = a.w + b.w;
    mx = mx > 0.f ? mx : mx * SLOPE;
    my = my > 0.f ? my : my * SLOPE;
    mz = mz > 0.f ? mz : mz * SLOPE;
    mw = mw > 0.f ? mw : mw * SLOPE;
    float s = mx * w.x + my * w.y + mz * w.z + mw * w.w;

    // butterfly reduction across the 8-lane group (wave32 cross-lane)
    s += __shfl_xor(s, 1, 8);
    s += __shfl_xor(s, 2, 8);
    s += __shfl_xor(s, 4, 8);

    if (q == 0) {
        earr[gid] = s;
        atomicMax(&emaxk[dst * NHEAD + h], fkey(s));
    }
}

// ---------------- exp(e - max) + segment sum ----------------
__global__ void k_expsum(const int* __restrict__ ei, int E, int Nn,
                         float* __restrict__ earr,
                         const unsigned* __restrict__ emaxk,
                         float* __restrict__ denom) {
    const int tid = blockIdx.x * blockDim.x + threadIdx.x;
    const int tot = (E + Nn) * NHEAD;
    if (tid >= tot) return;
    const int e = tid >> 2, h = tid & 3;
    const int dst = (e < E) ? ei[E + e] : (e - E);
    const float m  = funkey(emaxk[dst * NHEAD + h]);
    const float ex = __expf(earr[tid] - m);
    earr[tid] = ex;
    atomicAdd(&denom[dst * NHEAD + h], ex);
}

// ---------------- weighted scatter:  ACC[dst] += alpha * xl[src] ----------------
// 8 lanes per (edge, head): coalesced 128B read of xl[src] per group,
// 4 float atomics per lane to consecutive addresses.
__global__ void k_scatter(const int* __restrict__ ei, int E, int Nn,
                          const float* __restrict__ earr,
                          const float* __restrict__ denom,
                          const float* __restrict__ XL,
                          float* __restrict__ ACC) {
    const long long tid = (long long)blockIdx.x * blockDim.x + threadIdx.x;
    const long long tot = (long long)(E + Nn) * NHEAD * 8;
    if (tid >= tot) return;
    const int gid = (int)(tid >> 3);
    const int q   = (int)(tid & 7);
    const int e = gid >> 2, h = gid & 3;
    int src, dst;
    if (e < E) { src = ei[e]; dst = ei[E + e]; } else { src = dst = e - E; }

    const float alpha = earr[gid] / denom[dst * NHEAD + h];
    const float4 xv = *(const float4*)(XL + (size_t)src * HC + h * CHAN + q * 4);
    float* od = ACC + (size_t)dst * HC + h * CHAN + q * 4;
    atomicAdd(od + 0, alpha * xv.x);
    atomicAdd(od + 1, alpha * xv.y);
    atomicAdd(od + 2, alpha * xv.z);
    atomicAdd(od + 3, alpha * xv.w);
}

// ---------------- post: bias + relu (concat layers) ----------------
__global__ void k_post_concat(const float* __restrict__ ACC,
                              const float* __restrict__ bias,
                              float* __restrict__ hout, int Nn) {
    const int tid = blockIdx.x * blockDim.x + threadIdx.x;
    if (tid >= Nn * HC) return;
    const float v = ACC[tid] + bias[tid & (HC - 1)];
    hout[tid] = v > 0.f ? v : 0.f;
}

// ---------------- post: head mean + bias + relu (final layer) ----------------
__global__ void k_post_final(const float* __restrict__ ACC,
                             const float* __restrict__ bias,
                             float* __restrict__ hout, int Nn) {
    const int tid = blockIdx.x * blockDim.x + threadIdx.x;
    if (tid >= Nn * CHAN) return;
    const int n = tid >> 5, c = tid & 31;
    const size_t base = (size_t)n * HC + c;
    float v = 0.25f * (ACC[base] + ACC[base + 32] + ACC[base + 64] + ACC[base + 96]);
    v += bias[c];
    hout[tid] = v > 0.f ? v : 0.f;
}

// ---------------- global mean pool: one lane per (node, channel) ----------------
__global__ void k_pool(const float* __restrict__ hfin, const int* __restrict__ batch,
                       float* __restrict__ pooled, float* __restrict__ counts, int Nn) {
    const int tid = blockIdx.x * blockDim.x + threadIdx.x;
    if (tid >= Nn * CHAN) return;
    const int n = tid >> 5, c = tid & 31;
    const int b = batch[n];
    atomicAdd(&pooled[b * CHAN + c], hfin[(size_t)n * CHAN + c]);
    if (c == 0) atomicAdd(&counts[b], 1.0f);
}

// ---------------- FC head: relu(p@W1+b1)@W2+b2 ----------------
__global__ void k_fc(const float* __restrict__ pooled, const float* __restrict__ counts,
                     const float* __restrict__ W1, const float* __restrict__ b1,
                     const float* __restrict__ W2, const float* __restrict__ b2,
                     float* __restrict__ out, int G) {
    const int g = blockIdx.x * blockDim.x + threadIdx.x;
    if (g >= G) return;
    const float inv = 1.0f / fmaxf(counts[g], 1.0f);
    float p[CHAN], z[CHAN];
#pragma unroll
    for (int i = 0; i < CHAN; ++i) p[i] = pooled[g * CHAN + i] * inv;
    for (int j = 0; j < CHAN; ++j) {
        float s = b1[j];
        for (int i = 0; i < CHAN; ++i) s += p[i] * W1[i * CHAN + j];
        z[j] = s > 0.f ? s : 0.f;
    }
    for (int k = 0; k < NCLS; ++k) {
        float s = b2[k];
        for (int j = 0; j < CHAN; ++j) s += z[j] * W2[j * NCLS + k];
        out[g * NCLS + k] = s;
    }
}

// =======================================================================
extern "C" void kernel_launch(void* const* d_in, const int* in_sizes, int n_in,
                              void* d_out, int out_size, void* d_ws, size_t ws_size,
                              hipStream_t stream) {
    const float* x     = (const float*)d_in[0];
    const int*   ei    = (const int*)d_in[1];
    const int*   batch = (const int*)d_in[2];

    const int Nn = in_sizes[0] / F_IN;   // 20000
    const int E  = in_sizes[1] / 2;      // 320000
    const int G  = out_size / NCLS;      // 64
    const int Etot = E + Nn;             // with self-loops

    // workspace layout (256B aligned slices)
    char* w = (char*)d_ws;
    auto carve = [&](size_t bytes) -> void* {
        void* p = (void*)w;
        w += (bytes + 255) & ~(size_t)255;
        return p;
    };
    float*    ACC    = (float*)carve((size_t)Nn * HC * 4);
    unsigned* emaxk  = (unsigned*)carve((size_t)Nn * NHEAD * 4);
    float*    denom  = (float*)carve((size_t)Nn * NHEAD * 4);
    float*    XL     = (float*)carve((size_t)Nn * HC * 4);
    float*    XR     = (float*)carve((size_t)Nn * HC * 4);
    float*    hA     = (float*)carve((size_t)Nn * HC * 4);
    float*    hB     = (float*)carve((size_t)Nn * HC * 4);
    float*    earr   = (float*)carve((size_t)Etot * NHEAD * 4);
    float*    pooled = (float*)carve((size_t)G * CHAN * 4);
    float*    counts = (float*)carve((size_t)G * 4);
    (void)ws_size; (void)n_in;

    const long long edgeThreads8 = (long long)Etot * NHEAD * 8;
    const int edgeBlocks8 = (int)((edgeThreads8 + 255) / 256);
    const int edgeBlocks1 = (Etot * NHEAD + 255) / 256;
    const dim3 gemmGrid((Nn + 15) / 16, 2, 1);

    for (int i = 0; i < 4; ++i) {
        const float* Wl = (const float*)d_in[3 + 6 * i + 0];
        const float* bl = (const float*)d_in[3 + 6 * i + 1];
        const float* Wr = (const float*)d_in[3 + 6 * i + 2];
        const float* br = (const float*)d_in[3 + 6 * i + 3];
        const float* at = (const float*)d_in[3 + 6 * i + 4];
        const float* bs = (const float*)d_in[3 + 6 * i + 5];

        const float* hin = (i == 0) ? x : ((i & 1) ? hA : hB);
        float*       hout = (i & 1) ? hB : hA;
        const int din = (i == 0) ? F_IN : HC;

        // zero accumulators (emaxk zero == key(-inf) sentinel)
        k_zero<<<512, 256, 0, stream>>>(ACC, (long long)Nn * HC);
        k_zero<<<64, 256, 0, stream>>>((float*)emaxk, (long long)Nn * NHEAD);
        k_zero<<<64, 256, 0, stream>>>(denom, (long long)Nn * NHEAD);

        k_gemm<<<gemmGrid, 256, 0, stream>>>(hin, Wl, bl, Wr, br, XL, XR, Nn, din);
        k_score<<<edgeBlocks8, 256, 0, stream>>>(ei, E, Nn, XL, XR, at, earr, emaxk);
        k_expsum<<<edgeBlocks1, 256, 0, stream>>>(ei, E, Nn, earr, emaxk, denom);
        k_scatter<<<edgeBlocks8, 256, 0, stream>>>(ei, E, Nn, earr, denom, XL, ACC);

        if (i < 3) {
            k_post_concat<<<((Nn * HC) + 255) / 256, 256, 0, stream>>>(ACC, bs, hout, Nn);
        } else {
            k_post_final<<<((Nn * CHAN) + 255) / 256, 256, 0, stream>>>(ACC, bs, hout, Nn);
        }
    }

    const float* fcW1 = (const float*)d_in[27];
    const float* fcb1 = (const float*)d_in[28];
    const float* fcW2 = (const float*)d_in[29];
    const float* fcb2 = (const float*)d_in[30];
    const float* hfin = hB;   // layer 3 wrote hB

    k_zero<<<8, 256, 0, stream>>>(pooled, (long long)G * CHAN);
    k_zero<<<1, 256, 0, stream>>>(counts, (long long)G);
    k_pool<<<((Nn * CHAN) + 255) / 256, 256, 0, stream>>>(hfin, batch, pooled, counts, Nn);
    k_fc<<<(G + 63) / 64, 64, 0, stream>>>(pooled, counts, fcW1, fcb1, fcW2, fcb2,
                                           (float*)d_out, G);
}
